// PAM_Module_71451075936600
// MI455X (gfx1250) — compile-verified, matrix operand order checked by reading, not run on hardware
//
#include <hip/hip_runtime.h>
#include <math.h>

// ---------------------------------------------------------------------------
// PAM module for MI455X (gfx1250, wave32, WMMA, TDM).
//   B=4, C=512, H=W=64, HW=4096
//   out = gamma * (Bf @ softmax(Bf^T Bf)^T) + x,  Bf = Wb @ x
//
// Kernel 1: projection GEMM, f32 -> bf16 operands staged fragment-ready in
//           LDS, f32-accum v_wmma_f32_16x16x32_bf16.
// Kernel 2: flash-attention style fusion. KV tiles streamed by the Tensor
//           Data Mover into a double-buffered LDS tile (issue t+1 while
//           computing t; s_wait_tensorcnt + barrier publishes). Q/P/score-B
//           tiles stored pre-swizzled so every WMMA fragment is 2x
//           ds_load_b128 with zero repacking VALU.
// ---------------------------------------------------------------------------

typedef __attribute__((ext_vector_type(16))) __bf16 v16bf;
typedef __attribute__((ext_vector_type(8)))  float  v8f;
typedef __attribute__((ext_vector_type(4)))  unsigned int v4u;
typedef __attribute__((ext_vector_type(8)))  int v8i_t;
typedef __attribute__((ext_vector_type(4)))  int v4i_t;

#define C_DIM   512
#define HW_DIM  4096
#define B_DIM   4
#define NJT     (HW_DIM / 64)

#if __has_builtin(__builtin_amdgcn_tensor_load_to_lds) && \
    __has_builtin(__builtin_amdgcn_s_wait_tensorcnt)
#define HAVE_TDM 1
#else
#define HAVE_TDM 0
#warning "CDNA5 TDM builtins unavailable on this toolchain; using manual LDS staging fallback"
#endif

__device__ __forceinline__ __bf16 u2b(unsigned short u) {
    return __builtin_bit_cast(__bf16, u);
}
__device__ __forceinline__ unsigned short b2u(__bf16 b) {
    return __builtin_bit_cast(unsigned short, b);
}

// Fragment-ready swizzle (ISA 7.12.2, 16-bit A/B operands, K-chunk of 32):
//   kk in [0,32): lane-half h = (kk>>3)&1, slot = ((kk&16)>>1) | (kk&7)
// Row/col index (0..15) selects lane = idx + 16*h. Each lane's 16 elements
// are then contiguous -> one v16bf load = 2x ds_load_b128.
__device__ __forceinline__ int fr_h(int kk)    { return (kk >> 3) & 1; }
__device__ __forceinline__ int fr_slot(int kk) { return ((kk & 16) >> 1) | (kk & 7); }

__device__ __forceinline__ v16bf frag_ld(const __bf16* fr_base, int lane) {
    return *(const v16bf*)(fr_base + lane * 16);
}

// B fragment for the value GEMM: logical V[j][c] read from KV tile stored as
// KV[c][64]; bf16 pairs are contiguous along j -> lowers to 8x ds_load_b32.
__device__ __forceinline__ v16bf frag_B_T(const __bf16* kv, int c0, int j0, int lane) {
    int n = lane & 15, h = lane >> 4;
    v16bf b;
#pragma unroll
    for (int v = 0; v < 8; ++v) {
        int k = ((v >> 2) << 4) | (h << 3) | ((v & 3) << 1);
        b[2 * v]     = kv[(c0 + n) * 64 + j0 + k];
        b[2 * v + 1] = kv[(c0 + n) * 64 + j0 + k + 1];
    }
    return b;
}

#if HAVE_TDM
// Issue one Tensor-DMA: 512x64 bf16 tile (rows stride 4096 elems) -> LDS,
// row-major [c][j], data_size=2B. D# bitfields per ISA ch.8 (groups 0/1).
__device__ __forceinline__ void tdm_load_kv(const unsigned short* src,
                                            unsigned lds_off) {
    unsigned long long ga = (unsigned long long)(size_t)src;
    v4u g0 = {1u,                                   // count=1, no gather
              lds_off,                              // lds_addr
              (unsigned)ga,                         // global_addr[31:0]
              ((unsigned)(ga >> 32) & 0x01FFFFFFu)  // global_addr[56:32]
                  | 0x80000000u};                   // type=2 (image)
    v8i_t g1 = {(int)0x00010000u,       // data_size=1 (2 bytes)
                (int)(HW_DIM << 16),    // tensor_dim0 = 4096
                (int)(C_DIM << 16),     // tensor_dim1 = 512
                (int)(64 << 16),        // tile_dim0 = 64
                (int)C_DIM,             // tile_dim1 = 512, tile_dim2 = 0
                (int)HW_DIM,            // tensor_dim0_stride = 4096
                0, 0};                  // tensor_dim1_stride unused (2D)
    v4i_t z4 = {0, 0, 0, 0};
#if defined(__clang_major__) && __clang_major__ >= 23
    v8i_t z8 = {0, 0, 0, 0, 0, 0, 0, 0};
    __builtin_amdgcn_tensor_load_to_lds(g0, g1, z4, z4, z8, 0);
#else
    __builtin_amdgcn_tensor_load_to_lds(g0, g1, z4, z4, 0);
#endif
}
#endif

// ---------------------------------------------------------------------------
// Kernel 1: Bf[b][o][p] = sum_c Wb[o][c] * x[b][c][p]   (output stored bf16)
// Block: 256 thr (8 waves). Tile: M=128 (16 rows/wave), N=64, K staged by 32.
// ---------------------------------------------------------------------------
__global__ __launch_bounds__(256) void pam_proj_gemm(
    const float* __restrict__ x, const float* __restrict__ Wb,
    unsigned short* __restrict__ Bf) {
    __shared__ __align__(16) __bf16 Afr[8 * 32 * 16];  // [wstrip][lane][slot] 8KB
    __shared__ __align__(16) __bf16 Bfr[4 * 32 * 16];  // [ntile][lane][slot]  4KB

    const int tid = threadIdx.x, lane = tid & 31, w = tid >> 5;
    const int b = blockIdx.z, m0 = blockIdx.y * 128, n0 = blockIdx.x * 64;
    const float* xg = x + (size_t)b * C_DIM * HW_DIM;

    v8f acc[4] = {};

    for (int kc = 0; kc < C_DIM; kc += 32) {
        // Stage A tile 128x32 (Wb rows) -> fragment-ready bf16
        for (int it = tid; it < 1024; it += 256) {
            int row = it >> 3, col = (it & 7) << 2;        // col%4==0 -> same h,g
            float4 v4 = *(const float4*)(Wb + (size_t)(m0 + row) * C_DIM + kc + col);
            int ws = row >> 4, lm = (row & 15) + 16 * fr_h(col);
            __bf16* d = Afr + (ws * 32 + lm) * 16 + fr_slot(col);
            d[0] = (__bf16)v4.x; d[1] = (__bf16)v4.y;
            d[2] = (__bf16)v4.z; d[3] = (__bf16)v4.w;      // 4 contiguous slots
        }
        // Stage B tile 32x64 (x rows) -> fragment-ready bf16
        for (int it = tid; it < 512; it += 256) {
            int k = it >> 4, col = (it & 15) << 2;
            float4 v4 = *(const float4*)(xg + (size_t)(kc + k) * HW_DIM + n0 + col);
            int hh = fr_h(k), s = fr_slot(k), nt = col >> 4;
            __bf16* d = Bfr + (nt * 32 + ((col & 15) + 16 * hh)) * 16 + s;
            d[0]  = (__bf16)v4.x; d[16] = (__bf16)v4.y;    // lanes col..col+3
            d[32] = (__bf16)v4.z; d[48] = (__bf16)v4.w;
        }
        __syncthreads();

        v16bf af = frag_ld(Afr + w * 32 * 16, lane);       // 2x ds_load_b128
#pragma unroll
        for (int nt = 0; nt < 4; ++nt) {
            v16bf bf = frag_ld(Bfr + nt * 32 * 16, lane);
            acc[nt] = __builtin_amdgcn_wmma_f32_16x16x32_bf16(
                false, af, false, bf, (short)0, acc[nt], false, false);
        }
        __syncthreads();
    }

    const int h = lane >> 4, n = lane & 15;
#pragma unroll
    for (int nt = 0; nt < 4; ++nt) {
#pragma unroll
        for (int v = 0; v < 8; ++v) {
            int o = m0 + w * 16 + v + 8 * h;
            Bf[((size_t)b * C_DIM + o) * HW_DIM + n0 + nt * 16 + n] =
                b2u((__bf16)acc[nt][v]);
        }
    }
}

// ---------------------------------------------------------------------------
// Kernel 2: fused attention. One block per (b, 64-query tile).
// 8 waves: (mt 0..3) x (nh 0..1); each wave owns a 16x256 f32 E slice.
// KV tiles double-buffered: TDM load for tile t+1 issued while tile t is
// consumed; completion via s_wait_tensorcnt in the issuing wave + barrier.
// ---------------------------------------------------------------------------
__global__ __launch_bounds__(256) void pam_attention(
    const float* __restrict__ x, const unsigned short* __restrict__ BfU,
    const float* __restrict__ gammaP, float* __restrict__ out) {
    __shared__ __align__(16) __bf16 Qfr[4 * 16 * 32 * 16];  // frag-ready Q  64KB
    __shared__ __align__(16) __bf16 KV[2][C_DIM * 64];      // [c][j] x2    128KB
    __shared__ __align__(16) __bf16 KBfr[16 * 4 * 32 * 16]; // frag-ready K  64KB
    __shared__ __align__(16) float  Sf[64 * 64];            // raw scores    16KB
    __shared__ __align__(16) __bf16 Pfr[4 * 2 * 32 * 16];   // frag-ready P   8KB
    __shared__ float corrS[64];
    __shared__ float invl[64];

    const int tid = threadIdx.x, lane = tid & 31, w = tid >> 5;
    const int mt = w & 3, nh = w >> 2;
    const int h = lane >> 4, n = lane & 15;
    const int b = blockIdx.y, i0 = blockIdx.x * 64;
    const unsigned short* Bfb = BfU + (size_t)b * C_DIM * HW_DIM;

    // Stage Q^T once, pre-swizzled fragment-ready (coalesced global along i)
    for (int idx = tid; idx < 64 * C_DIM; idx += 256) {
        int i = idx & 63, c = idx >> 6;
        int kk = c & 31, kch = c >> 5;
        Qfr[(((i >> 4) * 16 + kch) * 32 + ((i & 15) + 16 * fr_h(kk))) * 16 +
            fr_slot(kk)] = u2b(Bfb[(size_t)c * HW_DIM + i0 + i]);
    }

    v8f Eacc[16] = {};
    float m_run = -INFINITY, l_run = 0.0f;

#if HAVE_TDM
    if (w == 0)  // prologue: start streaming tile 0
        tdm_load_kv(Bfb, (unsigned)(size_t)(void*)&KV[0][0]);
#endif

    for (int jt = 0; jt < NJT; ++jt) {
        const int j0 = jt * 64;
        const __bf16* KVc = &KV[jt & 1][0];

        // ---- Publish KV tile jt; kick off tile jt+1 ------------------------
#if HAVE_TDM
        if (w == 0) __builtin_amdgcn_s_wait_tensorcnt(0);
        __syncthreads();  // KV[jt&1] visible to all waves (also covers Qfr once)
        if (w == 0 && jt + 1 < NJT)
            tdm_load_kv(Bfb + j0 + 64, (unsigned)(size_t)(void*)&KV[(jt + 1) & 1][0]);
#else
        for (int idx = tid; idx < 8192; idx += 256) {
            int c = idx >> 4, col = (idx & 15) << 2;
            *(uint2*)(&KV[jt & 1][c * 64 + col]) =
                *(const uint2*)(Bfb + (size_t)c * HW_DIM + j0 + col);
        }
        __syncthreads();
#endif

        // ---- Reformat KV -> fragment-ready score-B tile KBfr ---------------
        for (int idx = tid; idx < 8192; idx += 256) {
            int c = idx >> 4, j4 = (idx & 15) << 2;        // 4 j's, same jtile
            union { uint2 u; unsigned short s[4]; } cv;
            cv.u = *(const uint2*)(KVc + c * 64 + j4);
            int kk = c & 31, kch = c >> 5, jtile = j4 >> 4;
            __bf16* d = KBfr + ((kch * 4 + jtile) * 32 +
                                ((j4 & 15) + 16 * fr_h(kk))) * 16 + fr_slot(kk);
            d[0]  = u2b(cv.s[0]); d[16] = u2b(cv.s[1]);
            d[32] = u2b(cv.s[2]); d[48] = u2b(cv.s[3]);
        }
        __syncthreads();

        // ---- Scores: S[mt*16..+16][nh*32..+32] = Q^T K ---------------------
        v8f sacc[2] = {};
        for (int kc = 0; kc < 16; ++kc) {
            v16bf af = frag_ld(Qfr + (mt * 16 + kc) * 32 * 16, lane);
#pragma unroll
            for (int nt = 0; nt < 2; ++nt) {
                v16bf bf = frag_ld(KBfr + (kc * 4 + nh * 2 + nt) * 32 * 16, lane);
                sacc[nt] = __builtin_amdgcn_wmma_f32_16x16x32_bf16(
                    false, af, false, bf, (short)0, sacc[nt], false, false);
            }
        }
#pragma unroll
        for (int nt = 0; nt < 2; ++nt)
#pragma unroll
            for (int v = 0; v < 8; ++v)
                Sf[(mt * 16 + v + 8 * h) * 64 + nh * 32 + nt * 16 + n] = sacc[nt][v];
        __syncthreads();

        // ---- Online softmax (exp on TRANS pipe co-executes with XDL) -------
        if (tid < 64) {
            const float* row = Sf + tid * 64;
            float tmax = row[0];
            for (int k = 1; k < 64; ++k) tmax = fmaxf(tmax, row[k]);
            float mnew = fmaxf(m_run, tmax);
            float corr = __expf(m_run - mnew);
            float ps = 0.0f;
            int pmt = tid >> 4, pm = tid & 15;
            for (int k = 0; k < 64; ++k) {
                float pv = __expf(row[k] - mnew);
                int kk = k & 31, jch = k >> 5;
                Pfr[((pmt * 2 + jch) * 32 + (pm + 16 * fr_h(kk))) * 16 + fr_slot(kk)] =
                    (__bf16)pv;
                ps += pv;
            }
            l_run = l_run * corr + ps;
            m_run = mnew;
            corrS[tid] = corr;
        }
        __syncthreads();

        // ---- Rescale E accumulators by exp(m_old - m_new) ------------------
        float f[8];
#pragma unroll
        for (int v = 0; v < 8; ++v) f[v] = corrS[mt * 16 + v + 8 * h];
#pragma unroll
        for (int t = 0; t < 16; ++t)
#pragma unroll
            for (int v = 0; v < 8; ++v) Eacc[t][v] *= f[v];

        // ---- Values: E[i][c] += P[i][j] * V[j][c] --------------------------
#pragma unroll
        for (int jk = 0; jk < 2; ++jk) {
            v16bf pf = frag_ld(Pfr + (mt * 2 + jk) * 32 * 16, lane);
#pragma unroll
            for (int t = 0; t < 16; ++t) {
                v16bf vf = frag_B_T(KVc, nh * 256 + t * 16, jk * 32, lane);
                Eacc[t] = __builtin_amdgcn_wmma_f32_16x16x32_bf16(
                    false, pf, false, vf, (short)0, Eacc[t], false, false);
            }
        }
        __syncthreads();  // protect Sf/Pfr/KBfr (and KV[(jt+1)&1] writer) reuse
    }

    if (tid < 64) invl[tid] = 1.0f / l_run;
    __syncthreads();

    const float gamma = gammaP[0];
    float il[8];
#pragma unroll
    for (int v = 0; v < 8; ++v) il[v] = invl[mt * 16 + v + 8 * h];

#pragma unroll
    for (int t = 0; t < 16; ++t) {
#pragma unroll
        for (int v = 0; v < 8; ++v) {
            int c = nh * 256 + t * 16 + n;
            int ir = i0 + mt * 16 + v + 8 * h;
            size_t idx = ((size_t)b * C_DIM + c) * HW_DIM + ir;
            out[idx] = gamma * Eacc[t][v] * il[v] + x[idx];
        }
    }
}

// ---------------------------------------------------------------------------
extern "C" void kernel_launch(void* const* d_in, const int* in_sizes, int n_in,
                              void* d_out, int out_size, void* d_ws, size_t ws_size,
                              hipStream_t stream) {
    const float* x     = (const float*)d_in[0];
    const float* Wb    = (const float*)d_in[1];
    const float* gamma = (const float*)d_in[2];
    float* out = (float*)d_out;
    unsigned short* Bf = (unsigned short*)d_ws;  // bf16 bits, B*C*HW elements

    dim3 g1(HW_DIM / 64, C_DIM / 128, B_DIM);    // 64 x 4 x 4
    pam_proj_gemm<<<g1, 256, 0, stream>>>(x, Wb, Bf);

    dim3 g2(HW_DIM / 64, B_DIM);                 // 64 x 4
    pam_attention<<<g2, 256, 0, stream>>>(x, Bf, gamma, out);
}
